// GINENet_39032662786178
// MI455X (gfx1250) — matrix operand order repaired and checked by reading, not compiled, not run on hardware
//
#include <hip/hip_runtime.h>
#include <hip/hip_bf16.h>

typedef __attribute__((ext_vector_type(16))) __bf16 v16bf;
typedef __attribute__((ext_vector_type(8)))  float  v8f;

#define N_NODES 50000

// ---------------------------------------------------------------------------
// Pack fp32 weight [K,256] (row-major, K=contraction dim) into bf16 WMMA
// B-fragment order: block (t = n-tile of 16 cols, c = k-chunk of 32 rows) at
// ((t*(K/32)+c)*512); within a block lane = hi*16 + n holds
// W[c*32 + kk][t*16 + n], kk = 8*hi + j (j < 8), 16 + 8*hi + (j - 8) (j >= 8) (32B per lane).
// ---------------------------------------------------------------------------
__global__ void pack_w_bf16(const float* __restrict__ W, __bf16* __restrict__ P, int K) {
    int tid = blockIdx.x * blockDim.x + threadIdx.x;
    int total = K * 256;
    if (tid >= total) return;
    int nkc  = K >> 5;
    int t    = tid / (nkc * 512);
    int rem  = tid % (nkc * 512);
    int c    = rem / 512;
    int lane = (rem & 511) >> 4;
    int j    = rem & 15;
    int hi   = lane >> 4;
    int n    = lane & 15;
    int k    = c * 32 + ((j < 8) ? (hi * 8 + j) : (16 + hi * 8 + (j - 8)));
    int col  = t * 16 + n;
    P[tid] = (__bf16)W[k * 256 + col];
}

__global__ void copy_f4(const float4* __restrict__ src, float4* __restrict__ dst, int n4) {
    int i = blockIdx.x * blockDim.x + threadIdx.x;
    if (i < n4) dst[i] = src[i];
}

// ---------------------------------------------------------------------------
// Edge phase: Acc[dst] += relu(X[src] + edge_attr @ ew + eb)
// Acc pre-initialized with node features (eps=0 => h = x + agg fused).
// Grid-stride loop over edge groups so the LDS staging of ew/eb is amortized
// across ~16-30 edge groups per block. Scatter-add uses guaranteed-native
// global_atomic_add_f32 (no-return, STOREcnt-tracked; s_endpgm drains).
// ---------------------------------------------------------------------------
template<int D>
__global__ __launch_bounds__(256)
void edge_msg(const float* __restrict__ X, const int* __restrict__ srcI,
              const int* __restrict__ dstI, const float* __restrict__ EA,
              const float* __restrict__ ew, const float* __restrict__ eb,
              float* __restrict__ Acc, int E) {
    constexpr int TPE = D / 4;        // threads per edge
    constexpr int BE  = 256 / TPE;    // edge groups per block pass
    constexpr int DP  = D + 4;        // pad to avoid LDS bank conflicts
    __shared__ float sEW[16 * DP];
    __shared__ float sEB[D];
    for (int idx = threadIdx.x; idx < 16 * D; idx += 256) {
        int k = idx / D, c = idx % D;
        sEW[k * DP + c] = ew[idx];
    }
    for (int idx = threadIdx.x; idx < D; idx += 256) sEB[idx] = eb[idx];
    __syncthreads();

    const int eL = threadIdx.x / TPE;
    const int li = threadIdx.x % TPE;
    const int c0 = li * 4;
    const int stride = gridDim.x * BE;

    for (int e = blockIdx.x * BE + eL; e < E; e += stride) {
        const int s = srcI[e];
        const int d = dstI[e];

        const float4* ea4 = (const float4*)(EA + (size_t)e * 16);
        float4 a0 = ea4[0], a1 = ea4[1], a2 = ea4[2], a3 = ea4[3];
        float ea[16] = {a0.x, a0.y, a0.z, a0.w, a1.x, a1.y, a1.z, a1.w,
                        a2.x, a2.y, a2.z, a2.w, a3.x, a3.y, a3.z, a3.w};

        float lin0 = sEB[c0 + 0], lin1 = sEB[c0 + 1];
        float lin2 = sEB[c0 + 2], lin3 = sEB[c0 + 3];
#pragma unroll
        for (int k = 0; k < 16; ++k) {
            const float av = ea[k];
            const float* wr = &sEW[k * DP + c0];
            lin0 = fmaf(av, wr[0], lin0);
            lin1 = fmaf(av, wr[1], lin1);
            lin2 = fmaf(av, wr[2], lin2);
            lin3 = fmaf(av, wr[3], lin3);
        }
        const float4 xv = *(const float4*)(X + (size_t)s * D + c0);
        float m0 = fmaxf(xv.x + lin0, 0.0f);
        float m1 = fmaxf(xv.y + lin1, 0.0f);
        float m2 = fmaxf(xv.z + lin2, 0.0f);
        float m3 = fmaxf(xv.w + lin3, 0.0f);

        float* ap = Acc + (size_t)d * D + c0;
        asm volatile(
            "global_atomic_add_f32 %0, %1, off\n\t"
            "global_atomic_add_f32 %0, %2, off offset:4\n\t"
            "global_atomic_add_f32 %0, %3, off offset:8\n\t"
            "global_atomic_add_f32 %0, %4, off offset:12"
            :: "v"(ap), "v"(m0), "v"(m1), "v"(m2), "v"(m3)
            : "memory");
    }
}

// ---------------------------------------------------------------------------
// Node MLP: Out = relu(H @ W1 + b1) @ W2 + b2, fused per 80-row block
// (50000 = 80 * 625, no tail).  256 threads = 8 waves; each wave owns two
// 16-col n-tiles across all five 16-row m-tiles (10 v8f accumulators), so
// every B-fragment load is reused by 5 WMMAs (5x less weight traffic).
// A-fragments from bank-padded LDS per the ISA 16-bit A layout.
// ---------------------------------------------------------------------------
template<int K>
__global__ __launch_bounds__(256)
void node_mlp(const float* __restrict__ Hin, const __bf16* __restrict__ W1,
              const float* __restrict__ b1, const __bf16* __restrict__ W2,
              const float* __restrict__ b2, float* __restrict__ Out) {
    constexpr int KP  = K + 8;
    constexpr int NKC = K / 32;
    constexpr int MT  = 5;                 // 5 m-tiles = 80 rows
    __shared__ __bf16 sH[80 * KP];
    __shared__ __bf16 sT[80 * 264];

    const int tid  = threadIdx.x;
    const int wave = tid >> 5;
    const int lane = tid & 31;
    const int hi   = lane >> 4;
    const int ln   = lane & 15;
    const int rowBase = blockIdx.x * 80;

    for (int idx = tid; idx < 80 * K; idx += 256) {
        int r = idx / K, k = idx % K;
        sH[r * KP + k] = (__bf16)Hin[(size_t)(rowBase + r) * K + k];
    }
    __syncthreads();

    const int t0 = wave * 2, t1 = wave * 2 + 1;
    const int n0 = t0 * 16 + ln, n1 = t1 * 16 + ln;

    // ---- GEMM1: T = relu(H @ W1 + b1)  (80 x 256) ----
    {
        v8f acc[MT][2] = {};
#pragma unroll
        for (int c = 0; c < NKC; ++c) {
            v16bf bf0 = *(const v16bf*)(W1 + ((size_t)(t0 * NKC + c) << 9) + lane * 16);
            v16bf bf1 = *(const v16bf*)(W1 + ((size_t)(t1 * NKC + c) << 9) + lane * 16);
#pragma unroll
            for (int m = 0; m < MT; ++m) {
                v16bf a;
                const __bf16* ar = &sH[(m * 16 + ln) * KP + c * 32 + hi * 8];
#pragma unroll
                for (int j = 0; j < 8; ++j) a[j] = ar[j];
#pragma unroll
                for (int j = 0; j < 8; ++j) a[8 + j] = ar[16 + j];
                acc[m][0] = __builtin_amdgcn_wmma_f32_16x16x32_bf16(
                    false, a, false, bf0, (short)0, acc[m][0], false, false);
                acc[m][1] = __builtin_amdgcn_wmma_f32_16x16x32_bf16(
                    false, a, false, bf1, (short)0, acc[m][1], false, false);
            }
        }
        const float bv0 = b1[n0], bv1 = b1[n1];
#pragma unroll
        for (int m = 0; m < MT; ++m) {
#pragma unroll
            for (int r = 0; r < 8; ++r) {
                const int row = m * 16 + hi * 8 + r;
                sT[row * 264 + n0] = (__bf16)fmaxf(acc[m][0][r] + bv0, 0.0f);
                sT[row * 264 + n1] = (__bf16)fmaxf(acc[m][1][r] + bv1, 0.0f);
            }
        }
    }
    __syncthreads();

    // ---- GEMM2: Out = T @ W2 + b2  (80 x 256, K2 = 256) ----
    {
        v8f acc[MT][2] = {};
#pragma unroll
        for (int c = 0; c < 8; ++c) {
            v16bf bf0 = *(const v16bf*)(W2 + ((size_t)(t0 * 8 + c) << 9) + lane * 16);
            v16bf bf1 = *(const v16bf*)(W2 + ((size_t)(t1 * 8 + c) << 9) + lane * 16);
#pragma unroll
            for (int m = 0; m < MT; ++m) {
                v16bf a;
                const __bf16* ar = &sT[(m * 16 + ln) * 264 + c * 32 + hi * 8];
#pragma unroll
                for (int j = 0; j < 8; ++j) a[j] = ar[j];
#pragma unroll
                for (int j = 0; j < 8; ++j) a[8 + j] = ar[16 + j];
                acc[m][0] = __builtin_amdgcn_wmma_f32_16x16x32_bf16(
                    false, a, false, bf0, (short)0, acc[m][0], false, false);
                acc[m][1] = __builtin_amdgcn_wmma_f32_16x16x32_bf16(
                    false, a, false, bf1, (short)0, acc[m][1], false, false);
            }
        }
        const float bv0 = b2[n0], bv1 = b2[n1];
#pragma unroll
        for (int m = 0; m < MT; ++m) {
#pragma unroll
            for (int r = 0; r < 8; ++r) {
                const size_t row = rowBase + m * 16 + hi * 8 + r;
                Out[row * 256 + n0] = acc[m][0][r] + bv0;
                Out[row * 256 + n1] = acc[m][1][r] + bv1;
            }
        }
    }
}

// ---------------------------------------------------------------------------
extern "C" void kernel_launch(void* const* d_in, const int* in_sizes, int n_in,
                              void* d_out, int out_size, void* d_ws, size_t ws_size,
                              hipStream_t stream) {
    const float* x    = (const float*)d_in[0];
    const int*   ei   = (const int*)d_in[1];
    const float* ea   = (const float*)d_in[2];
    const float* e1w  = (const float*)d_in[3];
    const float* e1b  = (const float*)d_in[4];
    const float* l11w = (const float*)d_in[5];
    const float* l11b = (const float*)d_in[6];
    const float* l12w = (const float*)d_in[7];
    const float* l12b = (const float*)d_in[8];
    const float* e2w  = (const float*)d_in[9];
    const float* e2b  = (const float*)d_in[10];
    const float* l21w = (const float*)d_in[11];
    const float* l21b = (const float*)d_in[12];
    const float* l22w = (const float*)d_in[13];
    const float* l22b = (const float*)d_in[14];

    const int E = in_sizes[1] / 2;
    const int* srcI = ei;
    const int* dstI = ei + E;

    // Workspace layout
    char* ws = (char*)d_ws;
    __bf16* pW11 = (__bf16*)(ws + 0);                       //  64 KB (128x256)
    __bf16* pW12 = (__bf16*)(ws + 65536);                   // 128 KB (256x256)
    __bf16* pW21 = (__bf16*)(ws + 65536 + 131072);          // 128 KB
    __bf16* pW22 = (__bf16*)(ws + 65536 + 2 * 131072);      // 128 KB
    float*  h1   = (float*)(ws + 524288);                             // N*128 f32
    float*  h2   = (float*)(ws + 524288 + (size_t)N_NODES * 128 * 4); // N*256 f32
    float*  out  = (float*)d_out;

    const int edgeBlocks = 6144;   // grid-stride; amortizes LDS weight staging

    // Pack weights to bf16 fragment order (runs every call; trivial cost)
    pack_w_bf16<<<(128 * 256 + 255) / 256, 256, 0, stream>>>(l11w, pW11, 128);
    pack_w_bf16<<<(256 * 256 + 255) / 256, 256, 0, stream>>>(l12w, pW12, 256);
    pack_w_bf16<<<(256 * 256 + 255) / 256, 256, 0, stream>>>(l21w, pW21, 256);
    pack_w_bf16<<<(256 * 256 + 255) / 256, 256, 0, stream>>>(l22w, pW22, 256);

    // ---- Layer 1 ----
    copy_f4<<<(N_NODES * 128 / 4 + 255) / 256, 256, 0, stream>>>(
        (const float4*)x, (float4*)h1, N_NODES * 128 / 4);            // h1 = x
    edge_msg<128><<<edgeBlocks, 256, 0, stream>>>(
        x, srcI, dstI, ea, e1w, e1b, h1, E);                           // h1 += agg
    node_mlp<128><<<N_NODES / 80, 256, 0, stream>>>(
        h1, pW11, l11b, pW12, l12b, out);                              // out1 -> d_out

    // ---- Layer 2 ----
    copy_f4<<<(N_NODES * 256 / 4 + 255) / 256, 256, 0, stream>>>(
        (const float4*)out, (float4*)h2, N_NODES * 256 / 4);          // h2 = out1
    edge_msg<256><<<edgeBlocks, 256, 0, stream>>>(
        out, srcI, dstI, ea, e2w, e2b, h2, E);                         // h2 += agg
    node_mlp<256><<<N_NODES / 80, 256, 0, stream>>>(
        h2, pW21, l21b, pW22, l22b, out);                              // final -> d_out
}
